// InterSliceAttention_70377334112557
// MI455X (gfx1250) — matrix operand order repaired and checked by reading, not compiled, not run on hardware
//
#include <hip/hip_runtime.h>

#define C_DIM 512
#define N_SP  1024
#define NHEAD 8
#define HD    64

typedef __attribute__((ext_vector_type(16))) _Float16 v16h;
typedef __attribute__((ext_vector_type(8)))  _Float16 v8h;
typedef __attribute__((ext_vector_type(8)))  float    v8f;

union FragU { unsigned int u[8]; v16h h; };

__device__ __forceinline__ v8f wmma16(v16h a, v16h b, v8f c) {
  return __builtin_amdgcn_wmma_f32_16x16x32_f16(false, a, false, b, (short)0, c, false, false);
}

// CDNA5 async tensor path: global -> LDS without VGPR round trip (ASYNCcnt tracked).
// LDS dest address = low 32 bits of the generic pointer (wave-relative LDS offset).
__device__ __forceinline__ void async_ld_b128(const _Float16* gsrc, const _Float16* ldst) {
  unsigned int l = (unsigned int)(unsigned long long)(uintptr_t)ldst;
  unsigned long long g = (unsigned long long)(uintptr_t)gsrc;
  asm volatile("global_load_async_to_lds_b128 %0, %1, off" :: "v"(l), "v"(g) : "memory");
}
__device__ __forceinline__ void wait_async0() {
  asm volatile("s_wait_asynccnt 0" ::: "memory");
}

// A fragment (16xK, f16): lane holds row M = row0 + (lane&15); half j -> K = j + (j>=8?8:0) + 8*hi
__device__ __forceinline__ v16h frag_a(const _Float16* base, int lda, int row0, int k0) {
  const int lane = threadIdx.x & 31;
  const int hi = lane >> 4;
  const unsigned int* q =
      (const unsigned int*)(base + (size_t)(row0 + (lane & 15)) * lda + k0 + hi * 8);
  FragU f;
#pragma unroll
  for (int t = 0; t < 8; ++t) f.u[t] = q[t + (t >= 4 ? 4 : 0)];
  return f.h;
}

// B fragment (Kx16, f16) from row-major [n][k] storage: lane holds col N = n0+(lane&15); K = j + (j>=8?8:0) + 8*hi
__device__ __forceinline__ v16h frag_b(const _Float16* base, int lda, int n0, int k0) {
  const int lane = threadIdx.x & 31;
  const int hi = lane >> 4;
  const unsigned int* q =
      (const unsigned int*)(base + (size_t)(n0 + (lane & 15)) * lda + k0 + hi * 8);
  FragU f;
#pragma unroll
  for (int t = 0; t < 8; ++t) f.u[t] = q[t + (t >= 4 ? 4 : 0)];
  return f.h;
}

// ---------------- Kernel 1: Q/K/V projections  Y = X @ W^T  (X from [C,HW] layout) ----------------
__global__ __launch_bounds__(128) void qkv_proj_kernel(
    const float* __restrict__ f_curr, const float* __restrict__ f_prev,
    const float* __restrict__ f_next, const float* __restrict__ Wq,
    const float* __restrict__ Wk, const float* __restrict__ Wv,
    _Float16* __restrict__ Qws, _Float16* __restrict__ Kws, _Float16* __restrict__ Vws)
{
  __shared__ _Float16 Ah[64 * 40];
  __shared__ _Float16 Bh[64 * 40];

  const int b  = blockIdx.z;
  const int rt = blockIdx.y;          // 0..15 Q, 16..47 K, 48..79 V
  const int c0 = blockIdx.x * 64;
  const int wave = threadIdx.x >> 5;
  const int lane = threadIdx.x & 31;
  const int hi = lane >> 4, lx = lane & 15;

  const float* src; const float* W; _Float16* dst;
  int srcL; size_t row0;
  if (rt < 16) {
    src = f_curr + (size_t)b * C_DIM * N_SP; W = Wq; dst = Qws;
    srcL = rt * 64; row0 = (size_t)b * 1024 + rt * 64;
  } else if (rt < 48) {
    const int m0 = (rt - 16) * 64;
    W = Wk; dst = Kws; row0 = (size_t)b * 2048 + m0;
    if (m0 < 1024) { src = f_prev + (size_t)b * C_DIM * N_SP; srcL = m0; }
    else           { src = f_next + (size_t)b * C_DIM * N_SP; srcL = m0 - 1024; }
  } else {
    const int m0 = (rt - 48) * 64;
    W = Wv; dst = Vws; row0 = (size_t)b * 2048 + m0;
    if (m0 < 1024) { src = f_prev + (size_t)b * C_DIM * N_SP; srcL = m0; }
    else           { src = f_next + (size_t)b * C_DIM * N_SP; srcL = m0 - 1024; }
  }

  v8f z = {};
  v8f acc[4];
#pragma unroll
  for (int m = 0; m < 4; ++m) acc[m] = z;

  for (int k0 = 0; k0 < C_DIM; k0 += 32) {
    __syncthreads();
#pragma unroll
    for (int i = 0; i < 16; ++i) {          // X tile: coalesced over spatial dim
      int idx = threadIdx.x + i * 128;
      int kk = idx >> 6, ll = idx & 63;
      Ah[ll * 40 + kk] = (_Float16)src[(size_t)(k0 + kk) * N_SP + srcL + ll];
    }
#pragma unroll
    for (int i = 0; i < 16; ++i) {          // W tile: rows contiguous
      int idx = threadIdx.x + i * 128;
      int nn = idx >> 5, kk = idx & 31;
      Bh[nn * 40 + kk] = (_Float16)W[(size_t)(c0 + nn) * C_DIM + k0 + kk];
    }
    __syncthreads();
    v16h bf = frag_b(Bh, 40, 16 * wave, 0);
#pragma unroll
    for (int m = 0; m < 4; ++m) {
      v16h af = frag_a(Ah, 40, 16 * m, 0);
      acc[m] = wmma16(af, bf, acc[m]);
    }
  }

#pragma unroll
  for (int m = 0; m < 4; ++m)
#pragma unroll
    for (int v = 0; v < 8; ++v)
      dst[(row0 + 16 * m + v + 8 * hi) * C_DIM + c0 + 16 * wave + lx] = (_Float16)acc[m][v];
}

// ---------------- Kernel 2: flash attention per (b, head, 64-query tile) ----------------
__global__ __launch_bounds__(128) void attn_kernel(
    const _Float16* __restrict__ Qws, const _Float16* __restrict__ Kws,
    const _Float16* __restrict__ Vws, _Float16* __restrict__ Ows)
{
  __shared__ __align__(16) _Float16 Qs[64 * 72];
  __shared__ __align__(16) _Float16 Ks[64 * 72];
  __shared__ __align__(16) _Float16 Vt[64 * 72];      // transposed: [d][m]
  __shared__ __align__(16) _Float16 Ps[4][16 * 72];   // per-wave P tile

  const int qt = blockIdx.x, h = blockIdx.y, b = blockIdx.z;
  const int l0 = qt * 64;
  const int wave = threadIdx.x >> 5;
  const int lane = threadIdx.x & 31;
  const int hi = lane >> 4, lx = lane & 15;

  const _Float16* Qb = Qws + ((size_t)b * 1024 + l0) * C_DIM + h * HD;
  const _Float16* Kb = Kws + ((size_t)b * 2048) * C_DIM + h * HD;
  const _Float16* Vb = Vws + ((size_t)b * 2048) * C_DIM + h * HD;

  // Q tile: async global -> LDS (16B chunks; row stride 144B in LDS, 1024B in global)
#pragma unroll
  for (int i = 0; i < 4; ++i) {
    int c = threadIdx.x + i * 128;          // 512 chunks = 64 rows x 8
    int l = c >> 3, d8 = (c & 7) * 8;
    async_ld_b128(Qb + (size_t)l * C_DIM + d8, &Qs[l * 72 + d8]);
  }

  v8f z = {};
  v8f Ot[4];
#pragma unroll
  for (int t = 0; t < 4; ++t) Ot[t] = z;
  float mi[8], li[8];
#pragma unroll
  for (int v = 0; v < 8; ++v) { mi[v] = -1e30f; li[v] = 0.0f; }

  for (int mb = 0; mb < 2048; mb += 64) {
    __syncthreads();
    // K tile: async global -> LDS
#pragma unroll
    for (int i = 0; i < 4; ++i) {
      int c = threadIdx.x + i * 128;
      int l = c >> 3, d8 = (c & 7) * 8;
      async_ld_b128(Kb + (size_t)(mb + l) * C_DIM + d8, &Ks[l * 72 + d8]);
    }
    // V tile: b128 vector loads, transpose on LDS store
#pragma unroll
    for (int i = 0; i < 4; ++i) {
      int c = threadIdx.x + i * 128;
      int m = c >> 3, d8 = (c & 7) * 8;
      v8h vv = *(const v8h*)(Vb + (size_t)(mb + m) * C_DIM + d8);
#pragma unroll
      for (int j = 0; j < 8; ++j) Vt[(d8 + j) * 72 + m] = vv[j];
    }
    wait_async0();
    __syncthreads();

    v16h qa0 = frag_a(Qs, 72, 16 * wave, 0);
    v16h qa1 = frag_a(Qs, 72, 16 * wave, 32);
    v8f S[4];
#pragma unroll
    for (int t = 0; t < 4; ++t) {
      v8f s = z;
      s = wmma16(qa0, frag_b(Ks, 72, 16 * t, 0), s);
      s = wmma16(qa1, frag_b(Ks, 72, 16 * t, 32), s);
      S[t] = s;
    }

    // online softmax: each output row lives in one 16-lane half-group at vgpr slot v
#pragma unroll
    for (int v = 0; v < 8; ++v) {
      float mnew = -1e30f;
#pragma unroll
      for (int t = 0; t < 4; ++t) {
        float x = S[t][v] * 0.125f;     // hd^-0.5
        S[t][v] = x;
        mnew = fmaxf(mnew, x);
      }
      mnew = fmaxf(mnew, __shfl_xor(mnew, 1, 32));
      mnew = fmaxf(mnew, __shfl_xor(mnew, 2, 32));
      mnew = fmaxf(mnew, __shfl_xor(mnew, 4, 32));
      mnew = fmaxf(mnew, __shfl_xor(mnew, 8, 32));
      float mN  = fmaxf(mi[v], mnew);
      float fsc = __expf(mi[v] - mN);
      mi[v] = mN;
      li[v] *= fsc;
#pragma unroll
      for (int t = 0; t < 4; ++t) Ot[t][v] *= fsc;
      float rs = 0.0f;
#pragma unroll
      for (int t = 0; t < 4; ++t) {
        float p = __expf(S[t][v] - mN);
        S[t][v] = p;
        rs += p;
      }
      rs += __shfl_xor(rs, 1, 32);
      rs += __shfl_xor(rs, 2, 32);
      rs += __shfl_xor(rs, 4, 32);
      rs += __shfl_xor(rs, 8, 32);
      li[v] += rs;
    }

    // D-layout -> A-layout reshape through per-wave LDS
#pragma unroll
    for (int t = 0; t < 4; ++t)
#pragma unroll
      for (int v = 0; v < 8; ++v)
        Ps[wave][(v + 8 * hi) * 72 + 16 * t + lx] = (_Float16)S[t][v];

    v16h pa0 = frag_a(Ps[wave], 72, 0, 0);
    v16h pa1 = frag_a(Ps[wave], 72, 0, 32);
#pragma unroll
    for (int t = 0; t < 4; ++t) {
      Ot[t] = wmma16(pa0, frag_b(Vt, 72, 16 * t, 0), Ot[t]);
      Ot[t] = wmma16(pa1, frag_b(Vt, 72, 16 * t, 32), Ot[t]);
    }
  }

  _Float16* Ob = Ows + ((size_t)b * 1024 + l0) * C_DIM + h * HD;
#pragma unroll
  for (int t = 0; t < 4; ++t)
#pragma unroll
    for (int v = 0; v < 8; ++v)
      Ob[(size_t)(16 * wave + v + 8 * hi) * C_DIM + 16 * t + lx] =
          (_Float16)(Ot[t][v] / li[v]);
}

// ---------------- Kernel 3: output projection  Y = O @ Wo^T (fp32 out) ----------------
__global__ __launch_bounds__(128) void out_proj_kernel(
    const _Float16* __restrict__ Ows, const float* __restrict__ Wo,
    float* __restrict__ Yws)
{
  __shared__ _Float16 Ah[64 * 40];
  __shared__ _Float16 Bh[64 * 40];

  const int b  = blockIdx.z;
  const int l0 = blockIdx.y * 64;
  const int c0 = blockIdx.x * 64;
  const int wave = threadIdx.x >> 5;
  const int lane = threadIdx.x & 31;
  const int hi = lane >> 4, lx = lane & 15;

  const _Float16* A = Ows + ((size_t)b * 1024 + l0) * C_DIM;

  v8f z = {};
  v8f acc[4];
#pragma unroll
  for (int m = 0; m < 4; ++m) acc[m] = z;

  for (int k0 = 0; k0 < C_DIM; k0 += 32) {
    __syncthreads();
#pragma unroll
    for (int i = 0; i < 16; ++i) {
      int idx = threadIdx.x + i * 128;
      int ll = idx >> 5, kk = idx & 31;
      Ah[ll * 40 + kk] = A[(size_t)ll * C_DIM + k0 + kk];
    }
#pragma unroll
    for (int i = 0; i < 16; ++i) {
      int idx = threadIdx.x + i * 128;
      int nn = idx >> 5, kk = idx & 31;
      Bh[nn * 40 + kk] = (_Float16)Wo[(size_t)(c0 + nn) * C_DIM + k0 + kk];
    }
    __syncthreads();
    v16h bf = frag_b(Bh, 40, 16 * wave, 0);
#pragma unroll
    for (int m = 0; m < 4; ++m) {
      v16h af = frag_a(Ah, 40, 16 * m, 0);
      acc[m] = wmma16(af, bf, acc[m]);
    }
  }

#pragma unroll
  for (int m = 0; m < 4; ++m)
#pragma unroll
    for (int v = 0; v < 8; ++v)
      Yws[((size_t)b * 1024 + l0 + 16 * m + v + 8 * hi) * C_DIM + c0 + 16 * wave + lx] =
          acc[m][v];
}

// ---------------- Kernel 4: residual + LayerNorm over C, write [B,C,H,W] ----------------
__global__ __launch_bounds__(256) void ln_kernel(
    const float* __restrict__ Yws, const float* __restrict__ f_curr,
    const float* __restrict__ gamma, const float* __restrict__ beta,
    float* __restrict__ out)
{
  const int row = blockIdx.x;            // b*1024 + l
  const int b = row >> 10, l = row & 1023;
  const float* y  = Yws + (size_t)row * C_DIM;
  const float* fc = f_curr + (size_t)b * C_DIM * N_SP + l;
  float* op = out + (size_t)b * C_DIM * N_SP + l;

  const int t = threadIdx.x;
  float v0 = y[t]       + fc[(size_t)t * N_SP];
  float v1 = y[t + 256] + fc[(size_t)(t + 256) * N_SP];

  __shared__ float red[8];
  float s = v0 + v1;
#pragma unroll
  for (int m = 16; m >= 1; m >>= 1) s += __shfl_xor(s, m, 32);
  if ((t & 31) == 0) red[t >> 5] = s;
  __syncthreads();
  float tot = 0.0f;
#pragma unroll
  for (int i = 0; i < 8; ++i) tot += red[i];
  const float mean = tot * (1.0f / 512.0f);

  float d0 = v0 - mean, d1 = v1 - mean;
  float s2 = d0 * d0 + d1 * d1;
#pragma unroll
  for (int m = 16; m >= 1; m >>= 1) s2 += __shfl_xor(s2, m, 32);
  __syncthreads();
  if ((t & 31) == 0) red[t >> 5] = s2;
  __syncthreads();
  float tot2 = 0.0f;
#pragma unroll
  for (int i = 0; i < 8; ++i) tot2 += red[i];
  const float inv = rsqrtf(tot2 * (1.0f / 512.0f) + 1e-5f);

  op[(size_t)t * N_SP]         = d0 * inv * gamma[t] + beta[t];
  op[(size_t)(t + 256) * N_SP] = d1 * inv * gamma[t + 256] + beta[t + 256];
}

extern "C" void kernel_launch(void* const* d_in, const int* in_sizes, int n_in,
                              void* d_out, int out_size, void* d_ws, size_t ws_size,
                              hipStream_t stream) {
  (void)in_sizes; (void)n_in; (void)out_size; (void)ws_size;
  const float* f_curr = (const float*)d_in[0];
  const float* f_prev = (const float*)d_in[1];
  const float* f_next = (const float*)d_in[2];
  const float* Wq     = (const float*)d_in[3];
  const float* Wk     = (const float*)d_in[4];
  const float* Wv     = (const float*)d_in[5];
  const float* Wo     = (const float*)d_in[6];
  const float* gamma  = (const float*)d_in[7];
  const float* beta   = (const float*)d_in[8];
  float* out = (float*)d_out;

  char* ws = (char*)d_ws;
  _Float16* Qws = (_Float16*)(ws);                        //  8 MB [B,1024,512] f16
  _Float16* Kws = (_Float16*)(ws + ((size_t)8  << 20));   // 16 MB [B,2048,512] f16
  _Float16* Vws = (_Float16*)(ws + ((size_t)24 << 20));   // 16 MB [B,2048,512] f16
  _Float16* Ows = (_Float16*)(ws + ((size_t)40 << 20));   //  8 MB [B,1024,512] f16
  float*    Yws = (float*)(ws);                           // 16 MB fp32; aliases dead Q/K

  dim3 g1(8, 80, 8);
  qkv_proj_kernel<<<g1, 128, 0, stream>>>(f_curr, f_prev, f_next, Wq, Wk, Wv,
                                          Qws, Kws, Vws);
  dim3 g2(16, NHEAD, 8);
  attn_kernel<<<g2, 128, 0, stream>>>(Qws, Kws, Vws, Ows);
  dim3 g3(8, 16, 8);
  out_proj_kernel<<<g3, 128, 0, stream>>>(Ows, Wo, Yws);
  ln_kernel<<<dim3(8 * 1024), 256, 0, stream>>>(Yws, f_curr, gamma, beta, out);
}
